// GaitnetActor_15212774162767
// MI455X (gfx1250) — matrix operand order repaired and checked by reading, not compiled, not run on hardware
//
#include <hip/hip_runtime.h>
#include <hip/hip_bf16.h>
#include <math.h>

typedef __attribute__((ext_vector_type(16))) _Float16 v16h;
typedef __attribute__((ext_vector_type(8)))  float    v8f;
typedef __attribute__((ext_vector_type(4)))  float    v4f;

#define BATCH   8192
#define OPT     16
#define OBS_DIM 150   // 22 shared + 16*8 unique

// Async staging of weight tiles via GLOBAL_LOAD_ASYNC_TO_LDS_B128 (CDNA5).
// Flip to 0 to fall back to plain load+ds_store staging.
#define USE_ASYNC_STAGE 1

// ---------------------------------------------------------------------------
// Fragment helpers (WMMA f16 16x16x32, wave32; A/B use this file's contiguous k order)
// A (16x32, f16): lane L -> row m = L&15, half h = L>>4.
//   element e (0..15): K = e + 16*h
// B (32x16, f16): lane L -> col n = L&15, half h = L>>4; element e: K = e + 16h
// C/D (16x16, f32): lane L -> col n = L&15; VGPR v -> row m = v + 8*(L>>4)
// ---------------------------------------------------------------------------

__device__ __forceinline__ v16h load_a_frag_lds(const _Float16* base, int strideH,
                                                int kbase, int lane) {
    const int m = lane & 15;
    const int h = lane >> 4;
    const _Float16* row = base + m * strideH + kbase + 16 * h;
    v16h a;
    ((v4f*)&a)[0] = *(const v4f*)(row);        // K = kbase+16h   .. +7
    ((v4f*)&a)[1] = *(const v4f*)(row + 8);    // K = kbase+16h+8 .. +7
    return a;
}

__device__ __forceinline__ v16h load_b_frag_gmem(const _Float16* W, int Kpad,
                                                 int n0, int kbase, int lane) {
    const int n = n0 + (lane & 15);
    const int h = lane >> 4;
    const _Float16* row = W + (size_t)n * Kpad + kbase + 16 * h;
    v16h b;
    ((v4f*)&b)[0] = *(const v4f*)(row);
    ((v4f*)&b)[1] = *(const v4f*)(row + 8);
    return b;
}

// B fragment from a block-shared, chunk-swizzled LDS weight tile.
// Tile layout: 16 rows x CH chunks (chunk = 8 halves / 16 B), chunk j of row n
// stored at chunk index (j & ~15) | ((j ^ n) & 15)  -> conflict-free reads.
__device__ __forceinline__ v16h load_b_frag_lds_swz(const _Float16* buf, int CH,
                                                    int kbase, int lane) {
    const int n = lane & 15;
    const int h = lane >> 4;
    const int j0 = (kbase >> 3) + 2 * h;
    const int j1 = j0 + 1;
    const int s0 = (j0 & ~15) | ((j0 ^ n) & 15);
    const int s1 = (j1 & ~15) | ((j1 ^ n) & 15);
    v16h b;
    ((v4f*)&b)[0] = *(const v4f*)(buf + (n * CH + s0) * 8);
    ((v4f*)&b)[1] = *(const v4f*)(buf + (n * CH + s1) * 8);
    return b;
}

// Stage one 16-row weight tile (rows n0..n0+15, CH chunks/row) into swizzled LDS.
__device__ __forceinline__ void stage_weights(const _Float16* __restrict__ W, int Kpad,
                                              int n0, _Float16* buf, int CH,
                                              int tid, int nthreads) {
    const int chunks = 16 * CH;
    for (int c = tid; c < chunks; c += nthreads) {
        int n = c / CH, j = c - n * CH;
        const _Float16* src = W + (size_t)(n0 + n) * Kpad + j * 8;
        _Float16* dst = buf + (n * CH + ((j & ~15) | ((j ^ n) & 15))) * 8;
#if USE_ASYNC_STAGE
        uint32_t dls = (uint32_t)(uintptr_t)dst;   // LDS byte offset (addr[31:0])
        uint64_t ga  = (uint64_t)(uintptr_t)src;
        asm volatile("global_load_async_to_lds_b128 %0, %1, off"
                     :: "v"(dls), "v"(ga) : "memory");
#else
        *(v4f*)dst = *(const v4f*)src;
#endif
    }
}

__device__ __forceinline__ void stage_wait() {
#if USE_ASYNC_STAGE
    asm volatile("s_wait_asynccnt 0x0" ::: "memory");
#endif
}

__device__ __forceinline__ v8f bias_acc(const float* bias, int n0, int lane) {
    const float bv = bias[n0 + (lane & 15)];
    v8f c;
#pragma unroll
    for (int v = 0; v < 8; ++v) c[v] = bv;
    return c;
}

__device__ __forceinline__ v8f relu8(v8f c) {
#pragma unroll
    for (int v = 0; v < 8; ++v) c[v] = fmaxf(c[v], 0.0f);
    return c;
}

__device__ __forceinline__ v8f wmma_f16(v16h a, v16h b, v8f c) {
    return __builtin_amdgcn_wmma_f32_16x16x32_f16(false, a, false, b,
                                                  (short)0, c, false, false);
}

__device__ __forceinline__ void store_c_lds_f16(_Float16* base, int strideH,
                                                int col0, v8f c, int lane) {
    const int n  = lane & 15;
    const int mh = (lane >> 4) * 8;
#pragma unroll
    for (int v = 0; v < 8; ++v)
        base[(mh + v) * strideH + col0 + n] = (_Float16)c[v];
}

// ---------------------------------------------------------------------------
// Weight packing: f32 [rows x ksrc] -> f16 [rows x kdst], zero pad K
// ---------------------------------------------------------------------------
__global__ void pack_f16_kernel(const float* __restrict__ src,
                                _Float16* __restrict__ dst,
                                int rows, int ksrc, int kdst) {
    int i = blockIdx.x * blockDim.x + threadIdx.x;
    int total = rows * kdst;
    if (i >= total) return;
    int r = i / kdst, k = i - r * kdst;
    dst[i] = (k < ksrc) ? (_Float16)src[(size_t)r * ksrc + k] : (_Float16)0.0f;
}

// ---------------------------------------------------------------------------
// Shared encoder: obs[:, :22] -> relu(W1 x + b1) -> relu(W2 h + b2) -> se f16
// ---------------------------------------------------------------------------
#define SE_A_STRIDE 40
#define SE_H_STRIDE 264
#define SE_WAVE_LDS (16 * SE_A_STRIDE + 16 * SE_H_STRIDE)   // halves

__global__ __launch_bounds__(128) void shared_encoder_kernel(
    const float* __restrict__ obs,
    const float* __restrict__ sb1, const float* __restrict__ sb2,
    const _Float16* __restrict__ sw1p, const _Float16* __restrict__ sw2p,
    _Float16* __restrict__ seOut) {
    extern __shared__ _Float16 smem[];
    const int lane = threadIdx.x & 31;
    const int wave = threadIdx.x >> 5;
    const int tile = blockIdx.x * 4 + wave;
    const int m0   = tile * 16;

    _Float16* sA = smem + wave * SE_WAVE_LDS;
    _Float16* sH = sA + 16 * SE_A_STRIDE;

    for (int idx = lane; idx < 16 * 32; idx += 32) {
        int r = idx >> 5, k = idx & 31;
        float v = (k < 22) ? obs[(size_t)(m0 + r) * OBS_DIM + k] : 0.0f;
        sA[r * SE_A_STRIDE + k] = (_Float16)v;
    }
    __syncthreads();

    v16h a0 = load_a_frag_lds(sA, SE_A_STRIDE, 0, lane);
#pragma unroll
    for (int ct = 0; ct < 16; ++ct) {
        v8f c = bias_acc(sb1, ct * 16, lane);
        v16h b = load_b_frag_gmem(sw1p, 32, ct * 16, 0, lane);
        c = relu8(wmma_f16(a0, b, c));
        store_c_lds_f16(sH, SE_H_STRIDE, ct * 16, c, lane);
    }
    __syncthreads();

    v16h af[8];
#pragma unroll
    for (int ks = 0; ks < 8; ++ks)
        af[ks] = load_a_frag_lds(sH, SE_H_STRIDE, ks * 32, lane);

    const int n  = lane & 15;
    const int mh = (lane >> 4) * 8;
#pragma unroll
    for (int ct = 0; ct < 16; ++ct) {
        v8f c = bias_acc(sb2, ct * 16, lane);
#pragma unroll
        for (int ks = 0; ks < 8; ++ks) {
            v16h b = load_b_frag_gmem(sw2p, 256, ct * 16, ks * 32, lane);
            c = wmma_f16(af[ks], b, c);
        }
        c = relu8(c);
#pragma unroll
        for (int v = 0; v < 8; ++v)
            seOut[(size_t)(m0 + mh + v) * 256 + ct * 16 + n] = (_Float16)c[v];
    }
}

// ---------------------------------------------------------------------------
// Unique encoder over B*OPT rows; tile == batch row b (16 options).
// ---------------------------------------------------------------------------
#define UE_A_STRIDE 40
#define UE_H_STRIDE 136
#define UE_WAVE_LDS (16 * UE_A_STRIDE + 16 + 16 * UE_H_STRIDE)

__global__ __launch_bounds__(128) void unique_encoder_kernel(
    const float* __restrict__ obs,
    const float* __restrict__ ub1, const float* __restrict__ ub2,
    const _Float16* __restrict__ uw1p, const _Float16* __restrict__ uw2p,
    const float* __restrict__ noop,
    _Float16* __restrict__ ueOut) {
    extern __shared__ _Float16 smem[];
    const int lane = threadIdx.x & 31;
    const int wave = threadIdx.x >> 5;
    const int b    = blockIdx.x * 4 + wave;

    _Float16* sA    = smem + wave * UE_WAVE_LDS;
    _Float16* sFlag = sA + 16 * UE_A_STRIDE;
    _Float16* sH    = sFlag + 16;

    for (int idx = lane; idx < 16 * 32; idx += 32) {
        int o = idx >> 5, k = idx & 31;
        float v = (k < 8) ? obs[(size_t)b * OBS_DIM + 22 + o * 8 + k] : 0.0f;
        sA[o * UE_A_STRIDE + k] = (_Float16)v;
    }
    if (lane < 16) {
        float u0 = obs[(size_t)b * OBS_DIM + 22 + lane * 8];
        sFlag[lane] = (u0 == 1.0f) ? (_Float16)1.0f : (_Float16)0.0f;
    }
    __syncthreads();

    v16h a0 = load_a_frag_lds(sA, UE_A_STRIDE, 0, lane);
#pragma unroll
    for (int ct = 0; ct < 8; ++ct) {
        v8f c = bias_acc(ub1, ct * 16, lane);
        v16h bw = load_b_frag_gmem(uw1p, 32, ct * 16, 0, lane);
        c = relu8(wmma_f16(a0, bw, c));
        store_c_lds_f16(sH, UE_H_STRIDE, ct * 16, c, lane);
    }
    __syncthreads();

    v16h af[4];
#pragma unroll
    for (int ks = 0; ks < 4; ++ks)
        af[ks] = load_a_frag_lds(sH, UE_H_STRIDE, ks * 32, lane);

    const int n  = lane & 15;
    const int mh = (lane >> 4) * 8;
#pragma unroll
    for (int ct = 0; ct < 8; ++ct) {
        v8f c = bias_acc(ub2, ct * 16, lane);
#pragma unroll
        for (int ks = 0; ks < 4; ++ks) {
            v16h bw = load_b_frag_gmem(uw2p, 128, ct * 16, ks * 32, lane);
            c = wmma_f16(af[ks], bw, c);
        }
        c = relu8(c);
        float nv = noop[ct * 16 + n];
#pragma unroll
        for (int v = 0; v < 8; ++v) {
            bool isNoOp = ((float)sFlag[mh + v]) != 0.0f;
            float val = isNoOp ? nv : c[v];
            ueOut[(size_t)(b * 16 + mh + v) * 128 + ct * 16 + n] = (_Float16)val;
        }
    }
}

// ---------------------------------------------------------------------------
// Trunk + heads. NT=2 batch tiles per wave; 4 waves/block share one
// double-buffered weight tile in LDS (staged with async-to-LDS copies) so each
// B fragment loaded from L2 once per block feeds 8 WMMAs.
// LDS: 4*58368 (activations) + 2*16384 (weight bufs) = 266 KB < 320 KB/WGP.
// ---------------------------------------------------------------------------
#define NT           2
#define TR_IN_STRIDE 392
#define TR_H_STRIDE  520
#define TR_T_STRIDE  264
#define TR_IN_TILE   (16 * TR_IN_STRIDE)      // 6272 halves
#define TR_H_TILE    (16 * TR_H_STRIDE)       // 8320 halves
#define TR_T_TILE    (16 * TR_T_STRIDE)       // 4224 halves
#define TR_WAVE_LDS  (NT * TR_IN_TILE + NT * TR_H_TILE)
#define TR_WAVES     4
#define TR_NTHREADS  (32 * TR_WAVES)
#define TR_WBUF_H    (16 * 512)               // halves per weight buffer
#define TR_TOTAL_LDS_H (TR_WAVES * TR_WAVE_LDS + 2 * TR_WBUF_H)

__global__ __launch_bounds__(TR_NTHREADS) void trunk_kernel(
    const _Float16* __restrict__ seF, const _Float16* __restrict__ ueF,
    const float* __restrict__ tb1, const float* __restrict__ tb2,
    const _Float16* __restrict__ tw1p, const _Float16* __restrict__ tw2p,
    const float* __restrict__ vw, const float* __restrict__ vb,
    const float* __restrict__ dw, const float* __restrict__ db,
    float* __restrict__ out) {
    extern __shared__ _Float16 smem[];
    const int tid  = threadIdx.x;
    const int lane = tid & 31;
    const int wave = tid >> 5;
    const int b0   = (blockIdx.x * TR_WAVES + wave) * NT;

    _Float16* sIn   = smem + wave * TR_WAVE_LDS;           // NT tiles; reused for t
    _Float16* sH    = sIn + NT * TR_IN_TILE;               // NT tiles
    _Float16* wbuf0 = smem + TR_WAVES * TR_WAVE_LDS;       // block-shared
    _Float16* wbuf1 = wbuf0 + TR_WBUF_H;

    // stage trunk_in for both tiles: [ se[b] (bcast) | ue[b*16+o] ] 16x384 f16
    for (int idx = lane; idx < NT * 16 * 384; idx += 32) {
        int t   = idx / (16 * 384);
        int rem = idx - t * (16 * 384);
        int r = rem / 384, c = rem - r * 384;
        int b = b0 + t;
        _Float16 v = (c < 256) ? seF[(size_t)b * 256 + c]
                               : ueF[(size_t)(b * 16 + r) * 128 + (c - 256)];
        sIn[t * TR_IN_TILE + r * TR_IN_STRIDE + c] = v;
    }
    // stage layer-1 weights for ct=0 (block-wide)
    stage_weights(tw1p, 384, 0, wbuf0, 48, tid, TR_NTHREADS);
    stage_wait();
    __syncthreads();

    // ---- layer 1: [16x384] @ [384x512] ----
    v16h af0[12];
#pragma unroll
    for (int ks = 0; ks < 12; ++ks)
        af0[ks] = load_a_frag_lds(sIn, TR_IN_STRIDE, ks * 32, lane);

#pragma unroll 1
    for (int ct = 0; ct < 32; ++ct) {
        _Float16* cur = (ct & 1) ? wbuf1 : wbuf0;
        _Float16* nxt = (ct & 1) ? wbuf0 : wbuf1;
        if (ct + 1 < 32)
            stage_weights(tw1p, 384, (ct + 1) * 16, nxt, 48, tid, TR_NTHREADS);
        v8f c0 = bias_acc(tb1, ct * 16, lane);
        v8f c1 = c0;
#pragma unroll
        for (int ks = 0; ks < 12; ++ks) {
            v16h bw = load_b_frag_lds_swz(cur, 48, ks * 32, lane);
            v16h a1 = load_a_frag_lds(sIn + TR_IN_TILE, TR_IN_STRIDE, ks * 32, lane);
            c0 = wmma_f16(af0[ks], bw, c0);
            c1 = wmma_f16(a1, bw, c1);
        }
        c0 = relu8(c0);
        c1 = relu8(c1);
        store_c_lds_f16(sH, TR_H_STRIDE, ct * 16, c0, lane);
        store_c_lds_f16(sH + TR_H_TILE, TR_H_STRIDE, ct * 16, c1, lane);
        stage_wait();
        __syncthreads();
    }

    // ---- layer 2: [16x512] @ [512x256] ----
    stage_weights(tw2p, 512, 0, wbuf0, 64, tid, TR_NTHREADS);

    v16h ag0[16];
#pragma unroll
    for (int ks = 0; ks < 16; ++ks)
        ag0[ks] = load_a_frag_lds(sH, TR_H_STRIDE, ks * 32, lane);
    stage_wait();
    __syncthreads();

#pragma unroll 1
    for (int ct = 0; ct < 16; ++ct) {
        _Float16* cur = (ct & 1) ? wbuf1 : wbuf0;
        _Float16* nxt = (ct & 1) ? wbuf0 : wbuf1;
        if (ct + 1 < 16)
            stage_weights(tw2p, 512, (ct + 1) * 16, nxt, 64, tid, TR_NTHREADS);
        v8f c0 = bias_acc(tb2, ct * 16, lane);
        v8f c1 = c0;
#pragma unroll
        for (int ks = 0; ks < 16; ++ks) {
            v16h bw = load_b_frag_lds_swz(cur, 64, ks * 32, lane);
            v16h a1 = load_a_frag_lds(sH + TR_H_TILE, TR_H_STRIDE, ks * 32, lane);
            c0 = wmma_f16(ag0[ks], bw, c0);
            c1 = wmma_f16(a1, bw, c1);
        }
        c0 = relu8(c0);
        c1 = relu8(c1);
        store_c_lds_f16(sIn, TR_T_STRIDE, ct * 16, c0, lane);   // reuse sIn
        store_c_lds_f16(sIn + TR_T_TILE, TR_T_STRIDE, ct * 16, c1, lane);
        stage_wait();
        __syncthreads();
    }

    // ---- heads: lanes 0-15 -> logits(row r); lanes 16-31 -> duration ----
    const int r    = lane & 15;
    const int head = lane >> 4;
    const float* w = head ? dw : vw;
#pragma unroll
    for (int t = 0; t < NT; ++t) {
        float acc = 0.0f;
        for (int c = 0; c < 256; ++c)
            acc += (float)sIn[t * TR_T_TILE + r * TR_T_STRIDE + c] * w[c];
        int b = b0 + t;
        if (head == 0) {
            out[(size_t)b * OPT + r] = acc + vb[0];
        } else {
            float x = acc + db[0];
            float s = 1.0f / (1.0f + __expf(-x));
            out[(size_t)BATCH * OPT + (size_t)b * OPT + r] = s * 0.4f + 0.1f;
        }
    }
}

// ---------------------------------------------------------------------------
// Launch
// ---------------------------------------------------------------------------
extern "C" void kernel_launch(void* const* d_in, const int* in_sizes, int n_in,
                              void* d_out, int out_size, void* d_ws, size_t ws_size,
                              hipStream_t stream) {
    const float* obs  = (const float*)d_in[0];
    const float* sw1  = (const float*)d_in[1];
    const float* sb1  = (const float*)d_in[2];
    const float* sw2  = (const float*)d_in[3];
    const float* sb2  = (const float*)d_in[4];
    const float* uw1  = (const float*)d_in[5];
    const float* ub1  = (const float*)d_in[6];
    const float* uw2  = (const float*)d_in[7];
    const float* ub2  = (const float*)d_in[8];
    const float* noop = (const float*)d_in[9];
    const float* tw1  = (const float*)d_in[10];
    const float* tb1  = (const float*)d_in[11];
    const float* tw2  = (const float*)d_in[12];
    const float* tb2  = (const float*)d_in[13];
    const float* vw   = (const float*)d_in[14];
    const float* vb   = (const float*)d_in[15];
    const float* dw   = (const float*)d_in[16];
    const float* db   = (const float*)d_in[17];
    float* out = (float*)d_out;

    _Float16* p    = (_Float16*)d_ws;
    _Float16* sw1p = p;              p += 256 * 32;
    _Float16* sw2p = p;              p += 256 * 256;
    _Float16* uw1p = p;              p += 128 * 32;
    _Float16* uw2p = p;              p += 128 * 128;
    _Float16* tw1p = p;              p += 512 * 384;
    _Float16* tw2p = p;              p += 256 * 512;
    _Float16* seF  = p;              p += (size_t)BATCH * 256;
    _Float16* ueF  = p;              // BATCH*16*128

    auto packs = [&](const float* s, _Float16* d, int r, int ks, int kd) {
        int total = r * kd;
        pack_f16_kernel<<<(total + 255) / 256, 256, 0, stream>>>(s, d, r, ks, kd);
    };
    packs(sw1, sw1p, 256, 22, 32);
    packs(sw2, sw2p, 256, 256, 256);
    packs(uw1, uw1p, 128, 8, 32);
    packs(uw2, uw2p, 128, 128, 128);
    packs(tw1, tw1p, 512, 384, 384);
    packs(tw2, tw2p, 256, 512, 512);

    shared_encoder_kernel<<<BATCH / (16 * 4), 128,
                            4 * SE_WAVE_LDS * sizeof(_Float16), stream>>>(
        obs, sb1, sb2, sw1p, sw2p, seF);

    unique_encoder_kernel<<<BATCH / 4, 128,
                            4 * UE_WAVE_LDS * sizeof(_Float16), stream>>>(
        obs, ub1, ub2, uw1p, uw2p, noop, ueF);

    trunk_kernel<<<BATCH / (TR_WAVES * NT), TR_NTHREADS,
                   TR_TOTAL_LDS_H * sizeof(_Float16), stream>>>(
        seF, ueF, tb1, tb2, tw1p, tw2p, vw, vb, dw, db, out);
}
